// QuanDense_6279242186830
// MI455X (gfx1250) — 8-bit matrix operand order repaired and checked by reading, compiled, not run on hardware
//
#include <hip/hip_runtime.h>
#include <hip/hip_bf16.h>
#include <float.h>

typedef __attribute__((ext_vector_type(8))) int v8i;

__device__ __forceinline__ unsigned enc_f32(float f) {
  unsigned u = __float_as_uint(f);
  return (u & 0x80000000u) ? ~u : (u | 0x80000000u);
}
__device__ __forceinline__ float dec_f32(unsigned e) {
  unsigned u = (e & 0x80000000u) ? (e ^ 0x80000000u) : ~e;
  return __uint_as_float(u);
}

__global__ void k_init(unsigned* mm) {
  if (threadIdx.x == 0) {
    mm[0] = 0xFFFFFFFFu;
    mm[1] = 0x00000000u;
    mm[2] = 0xFFFFFFFFu;
    mm[3] = 0x00000000u;
  }
}

__global__ void k_minmax(const float* __restrict__ p, long long n4,
                         unsigned* __restrict__ mnS, unsigned* __restrict__ mxS) {
  float mn = FLT_MAX, mx = -FLT_MAX;
  long long stride = (long long)gridDim.x * blockDim.x;
  for (long long i = (long long)blockIdx.x * blockDim.x + threadIdx.x; i < n4; i += stride) {
    float4 v = ((const float4*)p)[i];
    mn = fminf(mn, fminf(fminf(v.x, v.y), fminf(v.z, v.w)));
    mx = fmaxf(mx, fmaxf(fmaxf(v.x, v.y), fmaxf(v.z, v.w)));
  }
#pragma unroll
  for (int o = 16; o > 0; o >>= 1) {
    mn = fminf(mn, __shfl_xor(mn, o, 32));
    mx = fmaxf(mx, __shfl_xor(mx, o, 32));
  }
  if ((threadIdx.x & 31) == 0) {
    atomicMin(mnS, enc_f32(mn));
    atomicMax(mxS, enc_f32(mx));
  }
}

__global__ void k_quant_x(const float* __restrict__ x, unsigned char* __restrict__ qx,
                          const unsigned* __restrict__ mm, long long n4) {
  float mn = dec_f32(mm[0]);
  float inv = 255.0f / (dec_f32(mm[1]) - mn);
  long long stride = (long long)gridDim.x * blockDim.x;
  for (long long i = (long long)blockIdx.x * blockDim.x + threadIdx.x; i < n4; i += stride) {
    float4 v = ((const float4*)x)[i];
    unsigned q0 = (unsigned)__float2int_rn(fminf(255.0f, fmaxf(0.0f, (v.x - mn) * inv)));
    unsigned q1 = (unsigned)__float2int_rn(fminf(255.0f, fmaxf(0.0f, (v.y - mn) * inv)));
    unsigned q2 = (unsigned)__float2int_rn(fminf(255.0f, fmaxf(0.0f, (v.z - mn) * inv)));
    unsigned q3 = (unsigned)__float2int_rn(fminf(255.0f, fmaxf(0.0f, (v.w - mn) * inv)));
    ((unsigned*)qx)[i] = q0 | (q1 << 8) | (q2 << 16) | (q3 << 24);
  }
}

__global__ __launch_bounds__(256) void k_quant_wT(const float* __restrict__ W,
                                                  unsigned char* __restrict__ qwT,
                                                  const unsigned* __restrict__ mm,
                                                  int K, int N) {
  __shared__ unsigned char tile[32 * 33];
  float mn = dec_f32(mm[2]);
  float inv = 255.0f / (dec_f32(mm[3]) - mn);
  int n0 = blockIdx.x * 32;
  int k0 = blockIdx.y * 32;
  int t = threadIdx.x;
  int nl = t & 31;
  int kb = t >> 5;
#pragma unroll
  for (int i = 0; i < 4; i++) {
    int kl = kb * 4 + i;
    float v = W[(size_t)(k0 + kl) * N + n0 + nl];
    unsigned q = (unsigned)__float2int_rn(fminf(255.0f, fmaxf(0.0f, (v - mn) * inv)));
    tile[kl * 33 + nl] = (unsigned char)q;
  }
  __syncthreads();
  int nl2 = t >> 3;
  int dw = t & 7;
  unsigned u = 0;
#pragma unroll
  for (int j = 0; j < 4; j++)
    u |= ((unsigned)tile[(dw * 4 + j) * 33 + nl2]) << (8 * j);
  ((unsigned*)(qwT + (size_t)(n0 + nl2) * K + k0))[dw] = u;
}

__global__ __launch_bounds__(256) void k_rowsum(const unsigned char* __restrict__ q,
                                                float* __restrict__ out, int cols) {
  __shared__ float red[8];
  int row = blockIdx.x;
  int tid = threadIdx.x;
  const uint4* p = (const uint4*)(q + (size_t)row * cols);
  unsigned s = 0;
  for (int j = tid; j < cols / 16; j += 256) {
    uint4 v = p[j];
    unsigned a = v.x, b = v.y, c = v.z, d = v.w;
    s += (a & 0xff) + ((a >> 8) & 0xff) + ((a >> 16) & 0xff) + (a >> 24);
    s += (b & 0xff) + ((b >> 8) & 0xff) + ((b >> 16) & 0xff) + (b >> 24);
    s += (c & 0xff) + ((c >> 8) & 0xff) + ((c >> 16) & 0xff) + (c >> 24);
    s += (d & 0xff) + ((d >> 8) & 0xff) + ((d >> 16) & 0xff) + (d >> 24);
  }
  float f = (float)s;
#pragma unroll
  for (int o = 16; o > 0; o >>= 1) f += __shfl_xor(f, o, 32);
  if ((tid & 31) == 0) red[tid >> 5] = f;
  __syncthreads();
  if (tid == 0) {
    float tsum = 0.0f;
#pragma unroll
    for (int i = 0; i < 8; i++) tsum += red[i];
    out[row] = tsum;
  }
}

__global__ __launch_bounds__(256) void k_gemm_iu8(
    const unsigned char* __restrict__ qx,
    const unsigned char* __restrict__ qwT,
    const float* __restrict__ rowsumX,
    const float* __restrict__ colsumW,
    const float* __restrict__ bias,
    const unsigned* __restrict__ mm,
    float* __restrict__ out,
    int M, int N, int K) {
  __shared__ unsigned char lA[128 * 64];
  __shared__ unsigned char lB[128 * 64];

  const int tid = threadIdx.x;
  const int lane = tid & 31;
  const int laneHi = lane >> 4;
  const int lm = lane & 15;
  const int wave = tid >> 5;
  const int wmOff = (wave & 3) * 32;
  const int wnOff = (wave >> 2) * 64;

  const int mTile = blockIdx.y * 128;
  const int nTile = blockIdx.x * 128;

  v8i acc[2][4];
#pragma unroll
  for (int i = 0; i < 2; i++)
#pragma unroll
    for (int j = 0; j < 4; j++)
#pragma unroll
      for (int r = 0; r < 8; r++) acc[i][j][r] = 0;

  for (int k0 = 0; k0 < K; k0 += 64) {
#pragma unroll
    for (int i = 0; i < 2; i++) {
      int idx = tid + i * 256;
      int row = idx >> 2, seg = idx & 3;
      *(uint4*)(lA + row * 64 + seg * 16) =
          *(const uint4*)(qx + (size_t)(mTile + row) * K + k0 + seg * 16);
    }
#pragma unroll
    for (int i = 0; i < 2; i++) {
      int idx = tid + i * 256;
      int row = idx >> 2, seg = idx & 3;
      *(uint4*)(lB + row * 64 + seg * 16) =
          *(const uint4*)(qwT + (size_t)(nTile + row) * K + k0 + seg * 16);
    }
    __syncthreads();

    v8i a[2], b[4];
#pragma unroll
    for (int i = 0; i < 2; i++) {
      const unsigned char* pa = lA + (wmOff + i * 16 + lm) * 64;
      uint4 d0 = *(const uint4*)(pa + 0  + laneHi * 16);
      uint4 d1 = *(const uint4*)(pa + 32 + laneHi * 16);
      a[i][0] = d0.x; a[i][1] = d0.y; a[i][2] = d0.z; a[i][3] = d0.w;
      a[i][4] = d1.x; a[i][5] = d1.y; a[i][6] = d1.z; a[i][7] = d1.w;
    }
#pragma unroll
    for (int j = 0; j < 4; j++) {
      const unsigned char* pb = lB + (wnOff + j * 16 + lm) * 64 + laneHi * 16;
      uint4 e0 = *(const uint4*)(pb);
      uint4 e1 = *(const uint4*)(pb + 32);
      b[j][0] = e0.x; b[j][1] = e0.y; b[j][2] = e0.z; b[j][3] = e0.w;
      b[j][4] = e1.x; b[j][5] = e1.y; b[j][6] = e1.z; b[j][7] = e1.w;
    }

#pragma unroll
    for (int i = 0; i < 2; i++)
#pragma unroll
      for (int j = 0; j < 4; j++)
        acc[i][j] = __builtin_amdgcn_wmma_i32_16x16x64_iu8(
             false, a[i],  false, b[j], acc[i][j],
             false,  false);
    __syncthreads();
  }

  float xmn = dec_f32(mm[0]), xmx = dec_f32(mm[1]);
  float wmn = dec_f32(mm[2]), wmx = dec_f32(mm[3]);
  float sx = (xmx - xmn) * (1.0f / 255.0f);
  float sw = (wmx - wmn) * (1.0f / 255.0f);
  float c_xw = sx * sw;
  float c_cs = xmn * sw;
  float c_rs = wmn * sx;
  float c_0 = xmn * wmn * (float)K;

#pragma unroll
  for (int i = 0; i < 2; i++) {
#pragma unroll
    for (int j = 0; j < 4; j++) {
      int nCol = nTile + wnOff + j * 16 + lm;
      float cn = c_cs * colsumW[nCol] + bias[nCol] + c_0;
#pragma unroll
      for (int r = 0; r < 8; r++) {
        int mRow = mTile + wmOff + i * 16 + r + laneHi * 8;
        float v = c_xw * (float)acc[i][j][r] + c_rs * rowsumX[mRow] + cn;
        out[(size_t)mRow * N + nCol] = fmaxf(v, 0.0f);
      }
    }
  }
}

extern "C" void kernel_launch(void* const* d_in, const int* in_sizes, int n_in,
                              void* d_out, int out_size, void* d_ws, size_t ws_size,
                              hipStream_t stream) {
  const float* x = (const float*)d_in[0];
  const float* W = (const float*)d_in[1];
  const float* b = (const float*)d_in[2];
  float* out = (float*)d_out;

  const int Nunits = in_sizes[2];
  const int K = in_sizes[1] / Nunits;
  const int M = in_sizes[0] / K;

  unsigned char* ws = (unsigned char*)d_ws;
  unsigned* mm = (unsigned*)ws;
  unsigned char* qx = ws + 256;
  unsigned char* qwT = qx + (size_t)M * K;
  float* rowsumX = (float*)(qwT + (size_t)Nunits * K);
  float* colsumW = rowsumX + M;
  (void)ws_size; (void)n_in; (void)out_size;

  long long nx4 = (long long)M * K / 4;
  long long nw4 = (long long)K * Nunits / 4;

  k_init<<<1, 64, 0, stream>>>(mm);
  k_minmax<<<2048, 256, 0, stream>>>(x, nx4, mm + 0, mm + 1);
  k_minmax<<<2048, 256, 0, stream>>>(W, nw4, mm + 2, mm + 3);
  k_quant_x<<<2048, 256, 0, stream>>>(x, qx, mm, nx4);
  {
    dim3 gT(Nunits / 32, K / 32);
    k_quant_wT<<<gT, 256, 0, stream>>>(W, qwT, mm, K, Nunits);
  }
  k_rowsum<<<M, 256, 0, stream>>>(qx, rowsumX, K);
  k_rowsum<<<Nunits, 256, 0, stream>>>(qwT, colsumW, K);
  {
    dim3 gG(Nunits / 128, M / 128);
    k_gemm_iu8<<<gG, 256, 0, stream>>>(qx, qwT, rowsumX, colsumW, b, mm, out,
                                       M, Nunits, K);
  }
}
